// GraphSpace_39204461478500
// MI455X (gfx1250) — compile-verified, matrix operand order checked by reading, not run on hardware
//
#include <hip/hip_runtime.h>
#include <hip/hip_bf16.h>

// ---------------------------------------------------------------------------
// 2-layer GCN for MI455X (gfx1250, wave32).
//   layer: h = X @ W (bf16 WMMA, f32 accum); out = D^-1/2 A D^-1/2 h + b
// N=50000, E=800000, INP=HID=128, OUP=64.
// Edge phase is the roofline cost (~1.6GB of L2-resident gathers + f32
// atomics); GEMMs (2.5 GFLOP) run on v_wmma_f32_16x16x32_bf16 with packed
// fragment-major weights (clause'd global_load_b128) and LDS-staged A tiles.
// ---------------------------------------------------------------------------

typedef __attribute__((ext_vector_type(16))) __bf16 v16bf;
typedef __attribute__((ext_vector_type(8)))  __bf16 v8bf;
typedef __attribute__((ext_vector_type(8)))  float  v8f;

// ----------------------------- small utilities -----------------------------

__global__ void k_fill_f32(float* __restrict__ p, float v, int n) {
    int i = blockIdx.x * blockDim.x + threadIdx.x;
    if (i < n) p[i] = v;
}

__global__ void k_degree(const int* __restrict__ col, float* __restrict__ deg, int e) {
    int i = blockIdx.x * blockDim.x + threadIdx.x;
    if (i < e) atomicAdd(&deg[col[i]], 1.0f);
}

__global__ void k_rsqrt_inplace(float* __restrict__ d, int n) {
    int i = blockIdx.x * blockDim.x + threadIdx.x;
    if (i < n) {
        float v = d[i];
        d[i] = (v > 0.0f) ? rsqrtf(v) : 0.0f;
    }
}

// Per-edge symmetric norm, computed once and reused by both layers' scatters.
__global__ void k_edge_norm(const int* __restrict__ row, const int* __restrict__ col,
                            const float* __restrict__ dinv, float* __restrict__ wn, int e) {
    int i = blockIdx.x * blockDim.x + threadIdx.x;
    if (i < e) wn[i] = dinv[row[i]] * dinv[col[i]];
}

// Pre-pack W (K=128 x NCOLS fp32) into fragment-major bf16:
//   Bpk[((ct*4 + ks)*32 + lane)*16 + j] = W[k][ct*16 + (lane&15)]
//   with k = ks*32 + (lane>>4)*8 + (j<8 ? j : j+8)   (ISA 16-bit B layout)
// so each lane's 16 fragment elements are 32 contiguous bytes.
template <int NCOLS>
__global__ void k_pack_B(const float* __restrict__ W, __bf16* __restrict__ Bpk) {
    constexpr int TILES_N = NCOLS / 16;
    constexpr int TOTAL   = TILES_N * 4 * 32 * 16;  // == 128 * NCOLS
    int i = blockIdx.x * blockDim.x + threadIdx.x;
    if (i >= TOTAL) return;
    const int j    = i & 15;
    const int lane = (i >> 4) & 31;
    const int ks   = (i >> 9) & 3;
    const int ct   = i >> 11;
    const int n    = ct * 16 + (lane & 15);
    const int k    = ks * 32 + (lane >> 4) * 8 + (j < 8 ? j : j + 8);
    Bpk[i] = (__bf16)W[k * NCOLS + n];
}

// ------------------------------ WMMA GEMM ----------------------------------
// h[M x NCOLS] = A[M x 128] (f32 -> bf16) @ Bpk  ; fused epilogue also writes
// agg[i,f] = h[i,f]*dinv[i]^2 + bias[f]  (self-loop + bias init for scatter).
// Block = 256 threads = 8 waves; each wave owns one 16x16 tile.
template <int NCOLS>
__global__ void __launch_bounds__(256)
k_gemm_bf16_wmma(const float* __restrict__ A,
                 const __bf16* __restrict__ Bpk,
                 float* __restrict__ H,
                 float* __restrict__ AGG,
                 const float* __restrict__ dinv,
                 const float* __restrict__ bias,
                 int M) {
    constexpr int K        = 128;
    constexpr int TILES_N  = NCOLS / 16;   // 8 or 4
    constexpr int ROWTILES = 8 / TILES_N;  // 1 or 2
    constexpr int ROWS     = 16 * ROWTILES;
    constexpr int LDA      = K + 8;        // 136 elems (272B rows, 16B aligned)

    __shared__ __bf16 As[ROWS * LDA];

    const int tid      = threadIdx.x;
    const int lane     = tid & 31;
    const int wave     = tid >> 5;
    const int blockRow = blockIdx.x * ROWS;

    // Cooperative A-tile: fp32 global -> bf16 LDS, zero-pad OOB rows.
    for (int i = tid; i < ROWS * K; i += 256) {
        const int r  = i >> 7;
        const int k  = i & (K - 1);
        const int gr = blockRow + r;
        const float v = (gr < M) ? A[(size_t)gr * K + k] : 0.0f;
        As[r * LDA + k] = (__bf16)v;
    }
    __syncthreads();

    const int colTile = wave % TILES_N;
    const int rowTile = wave / TILES_N;
    const int mloc    = lane & 15;        // A-frag row within tile
    const int kbase   = (lane >> 4) * 8;  // K sub-offset per half-wave
    const int n       = colTile * 16 + (lane & 15);

    const __bf16* arow = &As[(rowTile * 16 + mloc) * LDA + kbase];
    const __bf16* bfrg = Bpk + ((size_t)(colTile * 4) * 32 + lane) * 16;

    v8f c = {};
    #pragma unroll
    for (int ks = 0; ks < 4; ++ks) {
        const int kk = ks * 32;
        // A fragment: two contiguous 16B LDS chunks (K {0..7,16..23} / {8..15,24..31})
        const v8bf alo = *(const v8bf*)(arow + kk);
        const v8bf ahi = *(const v8bf*)(arow + kk + 16);
        const v16bf a  = __builtin_shufflevector(alo, ahi,
                             0, 1, 2, 3, 4, 5, 6, 7, 8, 9, 10, 11, 12, 13, 14, 15);
        // B fragment: 32 contiguous bytes per lane from the packed weights
        const v16bf b = *(const v16bf*)(bfrg + (size_t)ks * 32 * 16);
        c = __builtin_amdgcn_wmma_f32_16x16x32_bf16(
                /*neg_a=*/false, a, /*neg_b=*/false, b,
                /*c_mod=*/(short)0, c, /*reuse_a=*/false, /*reuse_b=*/false);
    }

    // C/D layout: VGPR r -> M = (lane>=16 ? 8 : 0) + r ; N = lane&15.
    const int   mbase = blockRow + rowTile * 16 + (lane >> 4) * 8;
    const float bn    = bias[n];
    if (blockRow + ROWS <= M) {
        // Fast path (all but the last block): straight-line, clause-able stores.
        #pragma unroll
        for (int r = 0; r < 8; ++r) {
            const int   mg = mbase + r;
            const float hv = c[r];
            const float di = dinv[mg];
            const size_t o = (size_t)mg * NCOLS + n;
            H[o]   = hv;
            AGG[o] = hv * di * di + bn;   // fused self-loop + bias init
        }
    } else {
        #pragma unroll
        for (int r = 0; r < 8; ++r) {
            const int mg = mbase + r;
            if (mg < M) {
                const float hv = c[r];
                const float di = dinv[mg];
                const size_t o = (size_t)mg * NCOLS + n;
                H[o]   = hv;
                AGG[o] = hv * di * di + bn;
            }
        }
    }
}

// --------------------------- edge scatter-add ------------------------------
// One wave per edge: out[col, :] += h[row, :] * wn[edge].
// Lane owns 4 (F=128) or 2 (F=64) consecutive features -> b128/b64 gathers.
template <int F>
__global__ void __launch_bounds__(256)
k_edge_scatter(const int* __restrict__ row, const int* __restrict__ col,
               const float* __restrict__ wn,
               const float* __restrict__ h,
               float* __restrict__ out, int e) {
    constexpr int VEC = F / 32;  // 4 or 2
    const int lane = threadIdx.x & 31;
    const int wid  = (blockIdx.x * blockDim.x + threadIdx.x) >> 5;
    const int nw   = (gridDim.x * blockDim.x) >> 5;
    for (int ed = wid; ed < e; ed += nw) {
        const int s = row[ed];
        const int t = col[ed];
        const float w = wn[ed];
        const float* hs = h   + (size_t)s * F + lane * VEC;
        float*       ot = out + (size_t)t * F + lane * VEC;
        if (ed + nw < e)  // hint next source row toward cache
            __builtin_prefetch(h + (size_t)row[ed + nw] * F + lane * VEC, 0, 1);
        if constexpr (VEC == 4) {
            const float4 v = *(const float4*)hs;
            atomicAdd(&ot[0], v.x * w);
            atomicAdd(&ot[1], v.y * w);
            atomicAdd(&ot[2], v.z * w);
            atomicAdd(&ot[3], v.w * w);
        } else {
            const float2 v = *(const float2*)hs;
            atomicAdd(&ot[0], v.x * w);
            atomicAdd(&ot[1], v.y * w);
        }
    }
}

// ------------------------------- launcher ----------------------------------

extern "C" void kernel_launch(void* const* d_in, const int* in_sizes, int n_in,
                              void* d_out, int out_size, void* d_ws, size_t ws_size,
                              hipStream_t stream) {
    (void)in_sizes; (void)n_in; (void)out_size; (void)ws_size;

    constexpr int N   = 50000;
    constexpr int E   = 800000;
    constexpr int INP = 128;
    constexpr int HID = 128;
    constexpr int OUP = 64;

    const float* x  = (const float*)d_in[0];
    const int*   ei = (const int*)d_in[1];      // [2,E]
    const float* W1 = (const float*)d_in[2];
    const float* b1 = (const float*)d_in[3];
    const float* W2 = (const float*)d_in[4];
    const float* b2 = (const float*)d_in[5];
    const int* rowp = ei;
    const int* colp = ei + E;
    float* out = (float*)d_out;                 // [N, OUP]

    // workspace carve-out (256B aligned)
    char* ws = (char*)d_ws;
    size_t off = 0;
    auto carve = [&](size_t bytes) -> char* {
        char* p = ws + off;
        off = (off + bytes + 255) & ~(size_t)255;
        return p;
    };
    float*  dinv  = (float*)carve((size_t)N * 4);
    float*  wn    = (float*)carve((size_t)E * 4);
    float*  h1    = (float*)carve((size_t)N * HID * 4);
    float*  agg1  = (float*)carve((size_t)N * HID * 4);
    float*  h2    = (float*)carve((size_t)N * OUP * 4);
    __bf16* B1pk  = (__bf16*)carve((size_t)INP * HID * 2);
    __bf16* B2pk  = (__bf16*)carve((size_t)HID * OUP * 2);

    // ---- degrees -> dinv -> per-edge norms ----
    k_fill_f32<<<(N + 255) / 256, 256, 0, stream>>>(dinv, 1.0f, N);   // self-loop
    k_degree<<<(E + 255) / 256, 256, 0, stream>>>(colp, dinv, E);
    k_rsqrt_inplace<<<(N + 255) / 256, 256, 0, stream>>>(dinv, N);
    k_edge_norm<<<(E + 255) / 256, 256, 0, stream>>>(rowp, colp, dinv, wn, E);

    // ---- pack weights into WMMA fragment-major bf16 ----
    k_pack_B<HID><<<(INP * HID + 255) / 256, 256, 0, stream>>>(W1, B1pk);
    k_pack_B<OUP><<<(HID * OUP + 255) / 256, 256, 0, stream>>>(W2, B2pk);

    // ---- layer 1 ----
    k_gemm_bf16_wmma<HID><<<(N + 15) / 16, 256, 0, stream>>>(x, B1pk, h1, agg1, dinv, b1, N);
    k_edge_scatter<HID><<<(E + 7) / 8, 256, 0, stream>>>(rowp, colp, wn, h1, agg1, E);

    // ---- layer 2 (aggregate straight into d_out) ----
    k_gemm_bf16_wmma<OUP><<<(N + 31) / 32, 256, 0, stream>>>(agg1, B2pk, h2, out, dinv, b2, N);
    k_edge_scatter<OUP><<<(E + 7) / 8, 256, 0, stream>>>(rowp, colp, wn, h2, out, E);
}
